// ESRNN_68032281968650
// MI455X (gfx1250) — matrix operand order repaired and checked by reading, not compiled, not run on hardware
//
#include <hip/hip_runtime.h>
#include <hip/hip_bf16.h>
#include <cstddef>

// ---------- problem constants ----------
#define BQ    1024
#define TT    400
#define HGRU  30
#define GG    90          // 3*H
#define GW    386         // number of windows  (T - IN_W + 1)
#define PRED_T 358        // GW - OUT_W
#define OUTW  28
#define INW   15

// ---------- d_out offsets (floats, reference return order) ----------
#define OFF_PRED 0UL                 // (358,1024,28)
#define OFF_ACT  10264576UL          // (358,1024,15)
#define OFF_HP   15763456UL          // (1024,28)
#define OFF_HOLD 15792128UL          // (386,1024,28)
#define OFF_VAL  26859520UL          // (1024,28)
#define OFF_HAV  26888192UL          // (1024,28)
#define OFF_MSE  26916864UL          // scalar

// ---------- workspace offsets (floats) ----------
#define WS_SEXT 0UL                  // (1024,414)
#define WS_LEV  423936UL             // (1024,400)
#define WS_MSEP 833536UL             // (1024)
#define WS_WI   834560UL             // (386,1024,15)
#define WS_H1   6763520UL            // (386,1024,30)
#define WS_H2   18621440UL           // (386,1024,30)
#define WS_H3   30479360UL           // (386,1024,30)

typedef __attribute__((ext_vector_type(16))) __bf16 v16bf;
typedef __attribute__((ext_vector_type(8)))  float  v8f;

union BFV { v16bf v; unsigned short s[16]; };

__device__ __forceinline__ unsigned short f2bf(float f) {
  union { float f; unsigned u; } c; c.f = f;
  unsigned u = c.u;
  unsigned r = (u + 0x7FFFu + ((u >> 16) & 1u)) >> 16;   // round-to-nearest-even
  return (unsigned short)r;
}
__device__ __forceinline__ float sigm(float x) { return 1.f / (1.f + __expf(-x)); }

// ============================================================================
// Kernel A: exponential-smoothing scan, one thread per series.
// Produces s_ext (B,414), levels (B,400), per-series mse partial sums.
// ============================================================================
__global__ __launch_bounds__(256) void es_scan_kernel(
    const float* __restrict__ x, const float* __restrict__ alpha_raw,
    const float* __restrict__ betta_raw, const float* __restrict__ seas_raw,
    const int* __restrict__ indexes,
    float* __restrict__ s_ext, float* __restrict__ levels,
    float* __restrict__ msepart) {
  int b = blockIdx.x * blockDim.x + threadIdx.x;
  if (b >= BQ) return;
  int sid = indexes[b];
  float alpha = sigm(alpha_raw[sid]);
  float betta = sigm(betta_raw[sid]);
  float sp[7];
#pragma unroll
  for (int i = 0; i < 7; ++i) sp[i] = __expf(seas_raw[sid * 7 + i]);

  float* se  = s_ext  + (size_t)b * 414;
  float* lev = levels + (size_t)b * TT;
#pragma unroll
  for (int i = 0; i < 7; ++i) se[i] = sp[i];
  se[7] = sp[0];                       // s_init = [sp, sp[0]]

  // register queue = s_init[1:] = [sp1..sp6, sp0]
  float q0 = sp[1], q1 = sp[2], q2 = sp[3], q3 = sp[4], q4 = sp[5], q5 = sp[6], q6 = sp[0];

  const float* xr = x + (size_t)b * TT;
  float level = xr[0] / sp[0];
  lev[0] = level;

  float msesum = 0.f, prev = 0.f;
  int t = 1;
  auto step = [&](float& slot) {
    float xt = xr[t];
    float nl = alpha * (xt / slot) + (1.f - alpha) * level;
    float dl = __logf(nl / level);
    float ns = betta * (xt / nl) + (1.f - betta) * slot;
    slot = ns;
    level = nl;
    lev[t]    = nl;
    se[7 + t] = ns;                    // seas column 7+t
    if (t >= 2) msesum += dl - prev * prev;
    prev = dl;
    ++t;
  };
  for (int blk = 0; blk < 57; ++blk) { // 399 = 57 * 7, queue rotates in registers
    step(q0); step(q1); step(q2); step(q3); step(q4); step(q5); step(q6);
  }
#pragma unroll
  for (int i = 0; i < 7; ++i) se[407 + i] = se[400 + i];   // wrap: s_ext tail
  msepart[b] = msesum;
}

// ============================================================================
// Kernel B: build window_input (386,1024,15) and emit actual_values directly.
// wi[w,b,i] = x[b,w+i] / s_ext[b,w+i] / levels[b,w+14]
// ============================================================================
__global__ __launch_bounds__(256) void window_kernel(
    const float* __restrict__ x, const float* __restrict__ s_ext,
    const float* __restrict__ levels, float* __restrict__ wi,
    float* __restrict__ out_actual) {
  size_t gid = (size_t)blockIdx.x * blockDim.x + threadIdx.x;
  if (gid >= (size_t)GW * BQ * INW) return;
  int i = (int)(gid % INW);
  size_t r = gid / INW;
  int b = (int)(r % BQ);
  int w = (int)(r / BQ);
  float v = x[(size_t)b * TT + w + i] / s_ext[(size_t)b * 414 + w + i]
            / levels[(size_t)b * TT + w + 14];
  wi[((size_t)w * BQ + b) * INW + i] = v;
  if (w < PRED_T) out_actual[((size_t)w * BQ + b) * INW + i] = v;
}

// ============================================================================
// Kernel C: dilated GRU layer with bf16 WMMA (V_WMMA_F32_16X16X32_BF16).
//   chains = d*B independent recurrences; one wave owns 16 chains.
//   Per step per wave: 12 WMMAs (6 N-tiles x {input path, hidden path}).
//   block = 64 threads (2 waves); grid = d*B/32 blocks.
// A-matrix layout (16-bit 16x32):  lane<16: M=lane,    elems -> K = e  (same K order as B)
//                                  lane>=16: M=lane-16, elems -> K = 16 + e
// B-matrix layout (32x16):         lane<16: N=lane,    elems -> K = e
//                                  lane>=16: N=lane-16, elems -> K = 16 + e
// ============================================================================
__global__ __launch_bounds__(64) void gru_layer_kernel(
    const float* __restrict__ in,    // (Tn, B, C)
    float* __restrict__ out,         // (Tn, B, 30)
    const float* __restrict__ w_ih,  // (90, C)
    const float* __restrict__ w_hh,  // (90, 30)
    const float* __restrict__ b_ih, const float* __restrict__ b_hh,
    int Tn, int C, int d) {
  __shared__ float hbuf[2][16][32];
  __shared__ float gib[2][16][96];
  __shared__ float ghb[2][16][96];

  const int lane = threadIdx.x & 31;
  const int wv   = threadIdx.x >> 5;
  const int tile = blockIdx.x * 2 + wv;
  const int ch0  = tile * 16;
  const int j    = ch0 / BQ;          // dilation phase (uniform per wave)
  const int b0   = ch0 % BQ;
  const int m    = lane & 15;         // row / column-within-tile index
  const int kb   = (lane < 16) ? 0 : 16;  // B-frag K base
  const int ka   = (lane < 16) ? 0 : 8;   // unused; A frags also use kb
  const int roff = (lane < 16) ? 0 : 8;   // C/D row base

  // ---- pre-build weight (B) fragments and bias accumulator seeds ----
  BFV Bih[6], Bhh[6];
  float biasI[6], biasH[6];
#pragma unroll
  for (int n = 0; n < 6; ++n) {
    int col = n * 16 + m;             // gate column 0..95
#pragma unroll
    for (int e = 0; e < 16; ++e) {
      int k = kb + e;
      float vi = (col < GG && k < C)    ? w_ih[col * C + k]    : 0.f;
      float vh = (col < GG && k < HGRU) ? w_hh[col * HGRU + k] : 0.f;
      Bih[n].s[e] = f2bf(vi);
      Bhh[n].s[e] = f2bf(vh);
    }
    biasI[n] = (col < GG) ? b_ih[col] : 0.f;
    biasH[n] = (col < GG) ? b_hh[col] : 0.f;
  }

  for (int idx = lane; idx < 16 * 32; idx += 32)
    hbuf[wv][idx >> 5][idx & 31] = 0.f;
  __syncthreads();

  const int K = (Tn + d - 1) / d;
  for (int k = 0; k < K; ++k) {
    const int t = k * d + j;
    const bool valid = (t < Tn);      // wave-uniform

    // ---- A fragments: x_t (K-dim C, zero padded) and h (K-dim 30) ----
    BFV Ax, Ah;
    const float* xrow = in + ((size_t)t * BQ + (b0 + m)) * C;
#pragma unroll
    for (int e = 0; e < 16; ++e) {
      int kk = kb + e;
      float vx = 0.f;
      if (valid && kk < C) vx = xrow[kk];
      Ax.s[e] = f2bf(vx);
      float vh = (kk < HGRU) ? hbuf[wv][m][kk] : 0.f;
      Ah.s[e] = f2bf(vh);
    }

    // ---- 12 WMMAs: gi = x @ Wih^T + b_ih ; gh = h @ Whh^T + b_hh ----
#pragma unroll
    for (int n = 0; n < 6; ++n) {
      v8f aI = {biasI[n], biasI[n], biasI[n], biasI[n], biasI[n], biasI[n], biasI[n], biasI[n]};
      v8f aH = {biasH[n], biasH[n], biasH[n], biasH[n], biasH[n], biasH[n], biasH[n], biasH[n]};
      aI = __builtin_amdgcn_wmma_f32_16x16x32_bf16(false, Ax.v, false, Bih[n].v, (short)0, aI, false, false);
      aH = __builtin_amdgcn_wmma_f32_16x16x32_bf16(false, Ah.v, false, Bhh[n].v, (short)0, aH, false, false);
#pragma unroll
      for (int r = 0; r < 8; ++r) {
        gib[wv][roff + r][n * 16 + m] = aI[r];
        ghb[wv][roff + r][n * 16 + m] = aH[r];
      }
    }
    __syncthreads();

    // ---- gate nonlinearity + state update (480 elems / 32 lanes) ----
#pragma unroll
    for (int q = 0; q < 15; ++q) {
      int idx = q * 32 + lane;
      int row = idx / HGRU, col = idx % HGRU;
      float ir  = gib[wv][row][col],      hr = ghb[wv][row][col];
      float iz  = gib[wv][row][col + 30], hz = ghb[wv][row][col + 30];
      float inn = gib[wv][row][col + 60], hn = ghb[wv][row][col + 60];
      float r_ = sigm(ir + hr);
      float z_ = sigm(iz + hz);
      float n_ = tanhf(inn + r_ * hn);
      float h_ = hbuf[wv][row][col];
      float hnew = (1.f - z_) * n_ + z_ * h_;
      hbuf[wv][row][col] = hnew;
      if (valid) out[((size_t)t * BQ + (b0 + row)) * HGRU + col] = hnew;
    }
    __syncthreads();
  }
}

// ============================================================================
// Kernel D: head  y = tanh(h @ lin_w^T + lin_b) @ score_w^T + score_b
//   h = h1 + h2 (residual).  Writes holdout_output and its 358-step prefix
//   (== prediction_values, by the prefix property of dilated GRUs).
// ============================================================================
__global__ __launch_bounds__(256) void head_kernel(
    const float* __restrict__ h_a, const float* __restrict__ h_b,
    const float* __restrict__ lin_w, const float* __restrict__ lin_b,
    const float* __restrict__ score_w, const float* __restrict__ score_b,
    float* __restrict__ out_pred, float* __restrict__ out_hold) {
  __shared__ float lw[900], lb[30], sw[840], sb[28];
  int tid = threadIdx.x;
  for (int i = tid; i < 900; i += 256) lw[i] = lin_w[i];
  for (int i = tid; i < 840; i += 256) sw[i] = score_w[i];
  if (tid < 30) lb[tid] = lin_b[tid];
  if (tid < 28) sb[tid] = score_b[tid];
  __syncthreads();

  size_t row = (size_t)blockIdx.x * 256 + tid;      // row = w*1024 + b
  if (row >= (size_t)GW * BQ) return;
  float h[HGRU], u[HGRU];
#pragma unroll
  for (int c = 0; c < HGRU; ++c) h[c] = h_a[row * HGRU + c] + h_b[row * HGRU + c];
#pragma unroll 5
  for (int jj = 0; jj < HGRU; ++jj) {
    float a = lb[jj];
#pragma unroll
    for (int c = 0; c < HGRU; ++c) a += lw[jj * HGRU + c] * h[c];
    u[jj] = tanhf(a);
  }
  bool pre = row < (size_t)PRED_T * BQ;
#pragma unroll 4
  for (int kk = 0; kk < OUTW; ++kk) {
    float a = sb[kk];
#pragma unroll
    for (int c = 0; c < HGRU; ++c) a += sw[kk * HGRU + c] * u[c];
    out_hold[row * OUTW + kk] = a;
    if (pre) out_pred[row * OUTW + kk] = a;
  }
}

// ============================================================================
// Kernel E: holdout_prediction, val copy, hav_dn.
// ============================================================================
__global__ __launch_bounds__(256) void finalize_kernel(
    const float* __restrict__ val, const float* __restrict__ s_ext,
    const float* __restrict__ levels, const float* __restrict__ hold_out,
    float* __restrict__ out_hp, float* __restrict__ out_val,
    float* __restrict__ out_hav) {
  int gid = blockIdx.x * blockDim.x + threadIdx.x;
  if (gid >= BQ * OUTW) return;
  int b = gid / OUTW, kk = gid % OUTW;
  float hs = s_ext[(size_t)b * 414 + 386 + kk];     // s_ext[:, -28:]
  float lv = levels[(size_t)b * TT + TT - 1];
  float ho = hold_out[((size_t)(GW - 1) * BQ + b) * OUTW + kk];
  float rn = ho * hs * lv;
  out_hp[gid]  = (rn > 0.f) ? rn : 0.f;
  float v = val[gid];
  out_val[gid] = v;
  out_hav[gid] = v / hs / lv;
}

__global__ void mse_kernel(const float* __restrict__ msepart, float* __restrict__ out_mse) {
  __shared__ float s[256];
  int tid = threadIdx.x;
  float a = 0.f;
  for (int i = tid; i < BQ; i += 256) a += msepart[i];
  s[tid] = a;
  __syncthreads();
  for (int st = 128; st > 0; st >>= 1) {
    if (tid < st) s[tid] += s[tid + st];
    __syncthreads();
  }
  if (tid == 0) out_mse[0] = s[0] / (398.f * (float)BQ);
}

// ============================================================================
extern "C" void kernel_launch(void* const* d_in, const int* in_sizes, int n_in,
                              void* d_out, int out_size, void* d_ws, size_t ws_size,
                              hipStream_t stream) {
  (void)in_sizes; (void)n_in; (void)out_size; (void)ws_size;
  const float* x     = (const float*)d_in[0];
  const float* val   = (const float*)d_in[1];
  const float* a_raw = (const float*)d_in[2];
  const float* b_raw = (const float*)d_in[3];
  const float* s_raw = (const float*)d_in[4];
  const int*   idxs  = (const int*)d_in[5];
  const float* w_ih[4] = {(const float*)d_in[6],  (const float*)d_in[10],
                          (const float*)d_in[14], (const float*)d_in[18]};
  const float* w_hh[4] = {(const float*)d_in[7],  (const float*)d_in[11],
                          (const float*)d_in[15], (const float*)d_in[19]};
  const float* b_ih[4] = {(const float*)d_in[8],  (const float*)d_in[12],
                          (const float*)d_in[16], (const float*)d_in[20]};
  const float* b_hh[4] = {(const float*)d_in[9],  (const float*)d_in[13],
                          (const float*)d_in[17], (const float*)d_in[21]};
  const float* lin_w   = (const float*)d_in[22];
  const float* lin_b   = (const float*)d_in[23];
  const float* score_w = (const float*)d_in[24];
  const float* score_b = (const float*)d_in[25];

  float* out = (float*)d_out;
  float* ws  = (float*)d_ws;

  es_scan_kernel<<<4, 256, 0, stream>>>(x, a_raw, b_raw, s_raw, idxs,
                                        ws + WS_SEXT, ws + WS_LEV, ws + WS_MSEP);

  {
    size_t n = (size_t)GW * BQ * INW;
    window_kernel<<<(unsigned)((n + 255) / 256), 256, 0, stream>>>(
        x, ws + WS_SEXT, ws + WS_LEV, ws + WS_WI, out + OFF_ACT);
  }

  // GRU stack (chains = d*B, block handles 32 chains = 2 waves x 16)
  gru_layer_kernel<<<32, 64, 0, stream>>>(ws + WS_WI, ws + WS_H1,
      w_ih[0], w_hh[0], b_ih[0], b_hh[0], GW, INW, 1);
  gru_layer_kernel<<<64, 64, 0, stream>>>(ws + WS_H1, ws + WS_H2,
      w_ih[1], w_hh[1], b_ih[1], b_hh[1], GW, HGRU, 2);
  gru_layer_kernel<<<64, 64, 0, stream>>>(ws + WS_H2, ws + WS_H3,
      w_ih[2], w_hh[2], b_ih[2], b_hh[2], GW, HGRU, 2);
  gru_layer_kernel<<<192, 64, 0, stream>>>(ws + WS_H3, ws + WS_H1,
      w_ih[3], w_hh[3], b_ih[3], b_hh[3], GW, HGRU, 6);

  head_kernel<<<(GW * BQ + 255) / 256, 256, 0, stream>>>(
      ws + WS_H1, ws + WS_H2, lin_w, lin_b, score_w, score_b,
      out + OFF_PRED, out + OFF_HOLD);

  finalize_kernel<<<(BQ * OUTW + 255) / 256, 256, 0, stream>>>(
      val, ws + WS_SEXT, ws + WS_LEV, out + OFF_HOLD,
      out + OFF_HP, out + OFF_VAL, out + OFF_HAV);

  mse_kernel<<<1, 256, 0, stream>>>(ws + WS_MSEP, out + OFF_MSE);
}
